// FEFM_52888227283217
// MI455X (gfx1250) — matrix operand order repaired and checked by reading, not compiled, not run on hardware
//
#include <hip/hip_runtime.h>
#include <hip/hip_bf16.h>

// ---------------------------------------------------------------------------
// FEFM frequency-domain attention block for MI455X (gfx1250, wave32, WMMA).
// Matmul stages (conv1x1 GEMMs, Gram matrix, attention apply) run on
// V_WMMA_F32_16X16X32_F16 (f16 in, f32 accumulate). Conv weights are staged
// into LDS via the Tensor Data Mover (tensor_load_to_lds / s_wait_tensorcnt).
// DCT/IDCT (8x8), depthwise 3x3 and elementwise stages are bandwidth-bound
// VALU kernels.
// ---------------------------------------------------------------------------

typedef __attribute__((ext_vector_type(16))) _Float16 v16h;
typedef __attribute__((ext_vector_type(8)))  float    v8f;
typedef unsigned int u32;
typedef __attribute__((ext_vector_type(4)))  u32      u32x4;
typedef __attribute__((ext_vector_type(8)))  u32      u32x8;

#define HW 384
#define PLANE 147456LL          // 384*384  == L (= hh*ww*64, coincidentally equal)
#define CCH 64
#define NPART 192               // Gram partials per (b,head)

// ---------------------------------------------------------------------------
// conv1x1 as per-pixel GEMM:  out[b,co,p] = sum_ci w[co,ci] * in[b,ci,p]
// Block owns one 16-row co tile; the 16 x 64 f32 weight tile is staged into
// LDS once per workgroup by the TDM (D# built per ISA ch.8; lds_addr=0 since
// wlds is the only LDS object). Each wave computes 16(co) x 64(p): N-tile
// outer, K inner, A fragments preloaded, accumulator stored per N-tile to
// keep live ranges short (no scratch spills); K=64 unrolled -> 8 WMMAs.
// Fragment lane layouts used by this code (16-bit; contiguous K, not the ISA table order):
//   A 16x32: lane M = lane&15 ; elem e: K = 16*(lane>>4) + e
//   B 32x16: lane N = lane&15 ; elem e: K = 16*(lane>>4) + e
//   C/D f32: lane N = lane&15 ; vgpr v: M = v + 8*(lane>>4)
// ---------------------------------------------------------------------------
__global__ void __launch_bounds__(256, 1)
conv1x1_wmma_kernel(const float* __restrict__ in,
                    const float* __restrict__ w,
                    float* __restrict__ out,
                    int Cin /*==64*/, int Cout)
{
    __shared__ float wlds[16 * 64];          // sole LDS object -> offset 0
    const int lane = threadIdx.x & 31;
    const int wave = threadIdx.x >> 5;
    const int ln   = lane & 15;
    const int hi   = lane >> 4;
    const int co0  = blockIdx.y * 16;
    const int bb   = blockIdx.z;

    // --- TDM: stage rows [co0, co0+16) x Cin of w into LDS (wave 0 issues).
    if (wave == 0) {
        const unsigned long long ga =
            (unsigned long long)(const void*)(w + (long long)co0 * Cin);
        u32x4 g0;
        g0[0] = 1u;                                        // count=1 (valid user D#)
        g0[1] = 0u;                                        // lds_addr = 0
        g0[2] = (u32)(ga & 0xffffffffu);                   // global_addr[31:0]
        g0[3] = (u32)((ga >> 32) & 0x01ffffffu)            // global_addr[56:32]
              | 0x80000000u;                               // type = 2 ("image")
        u32x8 g1;
        g1[0] = 0x00020000u;                               // data_size=2 (4 bytes)
        g1[1] = ((u32)Cin & 0xffffu) << 16;                // tensor_dim0[15:0]
        g1[2] = (((u32)Cin >> 16) & 0xffffu)               // tensor_dim0[31:16]
              | (((u32)Cout & 0xffffu) << 16);             // tensor_dim1[15:0]
        g1[3] = (((u32)Cout >> 16) & 0xffffu)              // tensor_dim1[31:16]
              | (((u32)Cin & 0xffffu) << 16);              // tile_dim0 = Cin
        g1[4] = 16u;                                       // tile_dim1=16, tile_dim2=0
        g1[5] = (u32)Cin;                                  // tensor_dim0_stride[31:0]
        g1[6] = 0u;                                        // stride hi / dim1_stride lo
        g1[7] = 0u;
        asm volatile("tensor_load_to_lds %0, %1"
                     :: "s"(g0), "s"(g1)
                     : "memory");
        __builtin_amdgcn_s_wait_tensorcnt(0);
    }
    __syncthreads();

    const long long p0 = ((long long)blockIdx.x * 8 + wave) * 64;
    const float* __restrict__ inb  = in  + (long long)bb * Cin  * PLANE;
    float*       __restrict__ outb = out + (long long)bb * Cout * PLANE;

    // Preload both A fragments (K = 0..31 and 32..63) from LDS.
    v16h afrag[2];
#pragma unroll
    for (int kt = 0; kt < 2; ++kt)
#pragma unroll
        for (int e = 0; e < 16; ++e) {
            const int K = kt * 32 + 16 * hi + e;
            afrag[kt][e] = (_Float16)wlds[ln * 64 + K];
        }

#pragma unroll
    for (int nt = 0; nt < 4; ++nt) {
        const long long N = p0 + nt * 16 + ln;
        v8f acc = {};
#pragma unroll
        for (int kt = 0; kt < 2; ++kt) {
            v16h b;
#pragma unroll
            for (int e = 0; e < 16; ++e) {
                const int K = kt * 32 + 16 * hi + e;
                b[e] = (_Float16)inb[(long long)K * PLANE + N];
            }
            acc = __builtin_amdgcn_wmma_f32_16x16x32_f16(false, afrag[kt], false, b,
                                                         (short)0, acc, false, false);
        }
#pragma unroll
        for (int v = 0; v < 8; ++v) {
            const int M = co0 + v + 8 * hi;
            outb[(long long)M * PLANE + N] = acc[v];
        }
    }
}

// ---------------------------------------------------------------------------
// Depthwise 3x3, SAME padding. Optional PReLU post-op and accumulate.
// ---------------------------------------------------------------------------
__global__ void dw3_kernel(const float* __restrict__ in,
                           const float* __restrict__ w,
                           float* __restrict__ out,
                           int C, const float* __restrict__ alpha,
                           int accumulate, long long total)
{
    const long long idx = (long long)blockIdx.x * blockDim.x + threadIdx.x;
    if (idx >= total) return;
    const int x = (int)(idx % HW);
    const int y = (int)((idx / HW) % HW);
    const long long bc = idx / ((long long)HW * HW);
    const int ch = (int)(bc % C);
    const float* __restrict__ ib = in + bc * ((long long)HW * HW);
    const float* __restrict__ wc = w + ch * 9;
    float acc = 0.f;
#pragma unroll
    for (int dy = -1; dy <= 1; ++dy) {
        const int yy = y + dy;
        if (yy < 0 || yy >= HW) continue;
#pragma unroll
        for (int dx = -1; dx <= 1; ++dx) {
            const int xx = x + dx;
            if (xx < 0 || xx >= HW) continue;
            acc += ib[(long long)yy * HW + xx] * wc[(dy + 1) * 3 + (dx + 1)];
        }
    }
    if (alpha) { const float a = alpha[0]; acc = acc >= 0.f ? acc : a * acc; }
    if (accumulate) out[idx] += acc; else out[idx] = acc;
}

// ---------------------------------------------------------------------------
// 8x8 DCT-II / inverse per patch. Basis built in LDS:
//   C[i][j]  = 2*cos(pi*(2j+1)*i/16)
//   Ci[i][j] = C[j][i] * (j==0 ? 1/32 : 1/16)   (exact inverse of C)
// forward: image layout -> patch-flat layout ; inverse: patch-flat -> image.
// inCS/outCS = channels-per-batch stride (handles k living inside kv[2c]).
// 256 threads = 4 patches.
// ---------------------------------------------------------------------------
__global__ void dct8_kernel(const float* __restrict__ in,
                            float* __restrict__ out,
                            int inverse, int inCS, int outCS)
{
    __shared__ float Bm[64];
    __shared__ float X[4][64];
    const int t = threadIdx.x;
    if (t < 64) {
        const int i = t >> 3, j = t & 7;
        if (!inverse)
            Bm[t] = 2.f * cospif((float)((2 * j + 1) * i) / 16.f);
        else
            Bm[t] = 2.f * cospif((float)((2 * i + 1) * j) / 16.f) *
                    (j == 0 ? 0.03125f : 0.0625f);
    }
    const int s = t >> 6, e = t & 63;
    const long long pid = (long long)blockIdx.x * 4 + s;
    const int wb = (int)(pid % 48);
    const long long r1 = pid / 48;
    const int hb = (int)(r1 % 48);
    const long long r2 = r1 / 48;                 // bb*64 + c
    const int bb = (int)(r2 >> 6), c = (int)(r2 & 63);
    const long long inplane  = (long long)bb * inCS  + c;
    const long long outplane = (long long)bb * outCS + c;
    const int row = e >> 3, col = e & 7;
    const long long imgIn  = (inplane  * HW + hb * 8) * HW + wb * 8;
    const long long imgOut = (outplane * HW + hb * 8) * HW + wb * 8;
    const long long patIn  = inplane  * PLANE + (long long)(hb * 48 + wb) * 64;
    const long long patOut = outplane * PLANE + (long long)(hb * 48 + wb) * 64;

    X[s][e] = inverse ? in[patIn + e] : in[imgIn + (long long)row * HW + col];
    __syncthreads();

    float acc = 0.f;
#pragma unroll
    for (int mm = 0; mm < 8; ++mm) {
        float part = 0.f;
#pragma unroll
        for (int nn = 0; nn < 8; ++nn) part += Bm[col * 8 + nn] * X[s][mm * 8 + nn];
        acc += Bm[row * 8 + mm] * part;
    }
    if (inverse) out[imgOut + (long long)row * HW + col] = acc;
    else         out[patOut + e] = acc;
}

// out1 = qfft * kfft (elementwise)
__global__ void ewmul_kernel(const float* __restrict__ a,
                             const float* __restrict__ b,
                             float* __restrict__ o, long long total)
{
    const long long i = (long long)blockIdx.x * blockDim.x + threadIdx.x;
    if (i < total) o[i] = a[i] * b[i];
}

// Per-(b,c) L2 norms of qfft (idx 0..127) and kfft (idx 128..255).
__global__ void norm_kernel(const float* __restrict__ qf,
                            const float* __restrict__ kf,
                            float* __restrict__ norms)
{
    __shared__ float red[256];
    const int idx = blockIdx.x;
    const float* __restrict__ src = (idx < 128 ? qf : kf) + (long long)(idx & 127) * PLANE;
    float s = 0.f;
    for (long long i = threadIdx.x; i < PLANE; i += blockDim.x) {
        const float v = src[i]; s += v * v;
    }
    red[threadIdx.x] = s; __syncthreads();
    for (int o = 128; o > 0; o >>= 1) {
        if (threadIdx.x < o) red[threadIdx.x] += red[threadIdx.x + o];
        __syncthreads();
    }
    if (threadIdx.x == 0) norms[idx] = sqrtf(red[0]);
}

// ---------------------------------------------------------------------------
// Gram partials via WMMA: Gp[bh][p][ci][cj] = sum_{k in slice} qf[ci,k]*kf[cj,k]
// Each wave owns a contiguous K-slice of 768 (24 WMMA steps); fixed split so
// the later ordered reduce is deterministic (no float atomics).
// ---------------------------------------------------------------------------
__global__ void gram_wmma_kernel(const float* __restrict__ qf,
                                 const float* __restrict__ kf,
                                 float* __restrict__ gpart)
{
    const int lane = threadIdx.x & 31, wave = threadIdx.x >> 5;
    const int bh = blockIdx.y;                    // b*4 + head
    const int bb = bh >> 2, h = bh & 3;
    const int pidx = blockIdx.x * 8 + wave;       // 0..191
    const int ln = lane & 15, hi = lane >> 4;
    const float* __restrict__ qb = qf + ((long long)bb * CCH + h * 16) * PLANE;
    const float* __restrict__ kb = kf + ((long long)bb * CCH + h * 16) * PLANE;
    const long long k0 = (long long)pidx * 768;

    v8f acc = {};
    for (int s = 0; s < 24; ++s) {
        const long long kk = k0 + (long long)s * 32;
        __builtin_prefetch(&qb[(long long)ln * PLANE + kk + 64], 0, 1);
        v16h a, b;
#pragma unroll
        for (int e = 0; e < 16; ++e)
            a[e] = (_Float16)qb[(long long)ln * PLANE + kk + 16 * hi + e];
#pragma unroll
        for (int e = 0; e < 16; ++e)
            b[e] = (_Float16)kb[(long long)ln * PLANE + kk + 16 * hi + e];
        acc = __builtin_amdgcn_wmma_f32_16x16x32_f16(false, a, false, b,
                                                     (short)0, acc, false, false);
    }
    float* __restrict__ gp = gpart + ((long long)bh * NPART + pidx) * 256;
#pragma unroll
    for (int v = 0; v < 8; ++v) gp[(v + 8 * hi) * 16 + ln] = acc[v];
}

// Ordered reduce of partials + norm scaling + temperature + row softmax.
// One block per (b,head); thread = ci*16+cj.
__global__ void softmax_kernel(const float* __restrict__ gpart,
                               const float* __restrict__ norms,
                               const float* __restrict__ temp,
                               float* __restrict__ attn)
{
    __shared__ float m[16][17];
    const int bh = blockIdx.x;
    const int bb = bh >> 2, h = bh & 3;
    const int ci = threadIdx.x >> 4, cj = threadIdx.x & 15;
    const float* __restrict__ gp = gpart + (long long)bh * NPART * 256 + threadIdx.x;
    float s = 0.f;
    for (int p = 0; p < NPART; ++p) s += gp[(long long)p * 256];
    const float nq = fmaxf(norms[bb * CCH + h * 16 + ci], 1e-12f);
    const float nk = fmaxf(norms[128 + bb * CCH + h * 16 + cj], 1e-12f);
    const float val = s / (nq * nk) * temp[h];
    m[ci][cj] = val; __syncthreads();
    float mx = -3.4e38f;
    for (int j = 0; j < 16; ++j) mx = fmaxf(mx, m[ci][j]);
    const float e = __expf(val - mx);
    __syncthreads(); m[ci][cj] = e; __syncthreads();
    float ssum = 0.f;
    for (int j = 0; j < 16; ++j) ssum += m[ci][j];
    attn[bh * 256 + ci * 16 + cj] = e / ssum;
}

// o[bh,ci,l] = sum_cj attn[ci,cj] * out1[bh,cj,l]  — WMMA with K zero-padded
// from 16 to 32 (A cols 16..31 and B rows 16..31 are zero).
__global__ void attnapply_wmma_kernel(const float* __restrict__ attn,
                                      const float* __restrict__ out1,
                                      float* __restrict__ o)
{
    const int lane = threadIdx.x & 31, wave = threadIdx.x >> 5;
    const int bh = blockIdx.y;
    const int bb = bh >> 2, h = bh & 3;
    const long long l0 = ((long long)blockIdx.x * 8 + wave) * 16;
    const int ln = lane & 15, hi = lane >> 4;
    const float* __restrict__ src = out1 + ((long long)bb * CCH + h * 16) * PLANE;
    const float* __restrict__ at  = attn + bh * 256;

    v16h a, b;
#pragma unroll
    for (int e = 0; e < 16; ++e) {
        const int K = 16 * hi + e;
        a[e] = (K < 16) ? (_Float16)at[ln * 16 + K] : (_Float16)0.f;
    }
#pragma unroll
    for (int e = 0; e < 16; ++e) {
        const int K = 16 * hi + e;
        b[e] = (K < 16) ? (_Float16)src[(long long)K * PLANE + l0 + ln] : (_Float16)0.f;
    }
    v8f acc = {};
    acc = __builtin_amdgcn_wmma_f32_16x16x32_f16(false, a, false, b,
                                                 (short)0, acc, false, false);
    float* __restrict__ dst = o + ((long long)bb * CCH + h * 16) * PLANE;
#pragma unroll
    for (int v = 0; v < 8; ++v)
        dst[(long long)(v + 8 * hi) * PLANE + l0 + ln] = acc[v];
}

// lam = exp(sum lq1*lk1) - exp(sum lq2*lk2) + (0.8 - 0.6*exp(-0.3*6))
__global__ void lam_kernel(const float* __restrict__ lq1, const float* __restrict__ lk1,
                           const float* __restrict__ lq2, const float* __restrict__ lk2,
                           float* __restrict__ lam)
{
    __shared__ float s1[64], s2[64];
    const int t = threadIdx.x;
    s1[t] = lq1[t] * lk1[t];
    s2[t] = lq2[t] * lk2[t];
    __syncthreads();
    for (int o = 32; o > 0; o >>= 1) {
        if (t < o) { s1[t] += s1[t + o]; s2[t] += s2[t + o]; }
        __syncthreads();
    }
    if (t == 0)
        lam[0] = __expf(s1[0]) - __expf(s2[0]) + (0.8f - 0.6f * __expf(-1.8f));
}

// t1 = q * out_mid ; t2 = v * (1 - lam*out_mid)   (v lives inside kv[2c])
__global__ void t1t2_kernel(const float* __restrict__ q, const float* __restrict__ v,
                            const float* __restrict__ om, const float* __restrict__ lamp,
                            float* __restrict__ t1, float* __restrict__ t2,
                            long long total, long long vbstride)
{
    const long long i = (long long)blockIdx.x * blockDim.x + threadIdx.x;
    if (i >= total) return;
    const float lam = lamp[0];
    const long long per = (long long)CCH * PLANE;
    const long long bb = i / per, inner = i % per;
    const float ov = om[i];
    t1[i] = q[i] * ov;
    t2[i] = v[bb * vbstride + inner] * (1.f - lam * ov);
}

// ---------------------------------------------------------------------------
extern "C" void kernel_launch(void* const* d_in, const int* in_sizes, int n_in,
                              void* d_out, int out_size, void* d_ws, size_t ws_size,
                              hipStream_t stream) {
    (void)in_sizes; (void)n_in; (void)out_size; (void)ws_size;
    const float* x        = (const float*)d_in[0];
    const float* nir      = (const float*)d_in[1];
    const float* w_hidden = (const float*)d_in[2];
    const float* w_hid_nr = (const float*)d_in[3];
    const float* w_dw     = (const float*)d_in[4];
    const float* w_dw_nir = (const float*)d_in[5];
    const float* temp     = (const float*)d_in[6];
    const float* w_pmid   = (const float*)d_in[7];
    const float* w_pout   = (const float*)d_in[8];
    const float* w_p1a    = (const float*)d_in[9];
    const float* a_p1     = (const float*)d_in[10];
    const float* w_p1b    = (const float*)d_in[11];
    const float* w_p2a    = (const float*)d_in[12];
    const float* a_p2     = (const float*)d_in[13];
    const float* w_p2b    = (const float*)d_in[14];
    const float* lq1      = (const float*)d_in[15];
    const float* lk1      = (const float*)d_in[16];
    const float* lq2      = (const float*)d_in[17];
    const float* lk2      = (const float*)d_in[18];
    float* outp = (float*)d_out;

    const long long T = 2LL * CCH * PLANE;          // 18,874,368 floats
    float* ws = (float*)d_ws;
    float* A0 = ws;            // hidden -> qfft -> o -> t1 -> pooled stage
    float* A1 = A0 + T;        // q -> pooled (q dies at t1t2)
    float* A2 = A1 + T;        // nir_hidden (2T); later out1 = A2, oimg = A2+T
    float* A3 = A2 + 2 * T;    // kv (2T): k=A3 (b-stride 128 planes), v=A3+64*PLANE
    float* A4 = A3 + 2 * T;    // kfft -> outmid -> prelu temp
    float* A6 = A4 + T;        // t2
    float* gpart = A6 + T;                         // 8*192*256
    float* norms = gpart + 8LL * NPART * 256;      // 256
    float* attnb = norms + 256;                    // 8*256
    float* lamb  = attnb + 8 * 256;                // 1

    const dim3 blk(256);
    const dim3 gConv64(288, 4, 2);                 // (PLANE/64/8, Cout/16, b)
    const dim3 gConv128(288, 8, 2);
    const int  gElem64  = (int)(T / 256);          // 73728
    const int  gElem128 = (int)(2 * T / 256);      // 147456
    const int  gPatch   = 73728;                   // 294912 patches / 4

    // 1) hidden = conv1x1(x, w_hidden)
    conv1x1_wmma_kernel<<<gConv64, blk, 0, stream>>>(x, w_hidden, A0, 64, 64);
    // 2) nir_hidden = conv1x1(nir, w_hidden_nir)  [Cout=128]
    conv1x1_wmma_kernel<<<gConv128, blk, 0, stream>>>(nir, w_hid_nr, A2, 64, 128);
    // 3) q = dw3(hidden)
    dw3_kernel<<<gElem64, blk, 0, stream>>>(A0, w_dw, A1, 64, nullptr, 0, T);
    // 4) kv = dw3(nir_hidden)  [128 ch]
    dw3_kernel<<<gElem128, blk, 0, stream>>>(A2, w_dw_nir, A3, 128, nullptr, 0, 2 * T);
    // 5) qfft = dct2(patch(q))   (overwrites hidden)
    dct8_kernel<<<gPatch, blk, 0, stream>>>(A1, A0, 0, 64, 64);
    // 6) kfft = dct2(patch(k))   (k inside kv: per-batch channel stride 128)
    dct8_kernel<<<gPatch, blk, 0, stream>>>(A3, A4, 0, 128, 64);
    // 7) out1 = qfft * kfft      (reuses dead nir_hidden slot)
    ewmul_kernel<<<gElem64, blk, 0, stream>>>(A0, A4, A2, T);
    // 8) per-channel norms of qfft/kfft
    norm_kernel<<<256, blk, 0, stream>>>(A0, A4, norms);
    // 9) Gram partials (WMMA, fixed K split)
    gram_wmma_kernel<<<dim3(24, 8), blk, 0, stream>>>(A0, A4, gpart);
    // 10) ordered reduce + scale + softmax
    softmax_kernel<<<8, blk, 0, stream>>>(gpart, norms, temp, attnb);
    // 11) o = attn @ out1  (WMMA; overwrites qfft)
    attnapply_wmma_kernel<<<dim3(1152, 8), blk, 0, stream>>>(attnb, A2, A0);
    // 12) oimg = un-patch(idct2(o))
    dct8_kernel<<<gPatch, blk, 0, stream>>>(A0, A2 + T, 1, 64, 64);
    // 13) out_mid = conv1x1(oimg, w_proj_mid)   (overwrites kfft)
    conv1x1_wmma_kernel<<<gConv64, blk, 0, stream>>>(A2 + T, w_pmid, A4, 64, 64);
    // 14) lambda scalar
    lam_kernel<<<1, 64, 0, stream>>>(lq1, lk1, lq2, lk2, lamb);
    // 15) t1 = q*out_mid ; t2 = v*(1 - lam*out_mid)
    t1t2_kernel<<<gElem64, blk, 0, stream>>>(A1, A3 + 64 * PLANE, A4, lamb,
                                             A0, A6, T, 128LL * PLANE);
    // 16) pool branch 1: prelu(dw3(t1)) -> dw3 -> pooled (A1; q is dead)
    dw3_kernel<<<gElem64, blk, 0, stream>>>(A0, w_p1a, A4, 64, a_p1, 0, T);
    dw3_kernel<<<gElem64, blk, 0, stream>>>(A4, w_p1b, A1, 64, nullptr, 0, T);
    // 17) pool branch 2: prelu(dw3(t2)) -> dw3 (+accumulate into pooled)
    dw3_kernel<<<gElem64, blk, 0, stream>>>(A6, w_p2a, A4, 64, a_p2, 0, T);
    dw3_kernel<<<gElem64, blk, 0, stream>>>(A4, w_p2b, A1, 64, nullptr, 1, T);
    // 18) final: out = conv1x1(pooled, w_proj_out)
    conv1x1_wmma_kernel<<<gConv64, blk, 0, stream>>>(A1, w_pout, outp, 64, 64);
}
